// PairEmbedder_64690797412634
// MI455X (gfx1250) — matrix operand order repaired and checked by reading, not compiled, not run on hardware
//
#include <hip/hip_runtime.h>
#include <hip/hip_bf16.h>

// ---------------------------------------------------------------------------
// PairEmbedder fused kernel for gfx1250 (MI455X).
// Tokens N = B*L*L = 131072, C = 256, H = 4C = 1024.
// Per-token MLP runs through v_wmma_f32_16x16x32_f16 (fp32 accumulate).
// ---------------------------------------------------------------------------

typedef __attribute__((ext_vector_type(16))) _Float16 v16h;
typedef __attribute__((ext_vector_type(8)))  _Float16 v8h;
typedef __attribute__((ext_vector_type(8)))  float    v8f;

#define BATCH   2
#define SEQL    256
#define CDIM    256
#define NBINS   64
#define HDIM    1024
#define TM      64            // tokens per workgroup
#define KCHUNK  128           // H processed per chunk
#define SA_STRIDE 264         // halfs  (256 + 8 pad)
#define H1_STRIDE 136         // halfs  (128 + 8 pad)
#define SR_STRIDE 260         // floats (256 + 4 pad)

#define SRES_BYTES (TM * SR_STRIDE * 4)             // 66560
#define SA_BYTES   (TM * SA_STRIDE * 2)             // 33792
#define H1_BYTES   (TM * H1_STRIDE * 2)             // 17408
#define SMEM_BYTES (SRES_BYTES + SA_BYTES + H1_BYTES) // 117760 (< 320KB/WGP)

// A-fragment: lane m<16 holds row M=m, K = {0..7, 16..23}; lane>=16 holds
// K = {8..15, 24..31}.  Two 16-byte LDS loads per lane.
__device__ __forceinline__ v16h ld_afrag(const _Float16* p) {
    v8h lo = *(const v8h*)(p);
    v8h hi = *(const v8h*)(p + 16);
    v16h r;
#pragma unroll
    for (int i = 0; i < 8; ++i) { r[i] = lo[i]; r[i + 8] = hi[i]; }
    return r;
}

// B-fragment: lane n<16 holds col N=n, K = {0..7, 16..23}; lane>=16 holds
// K = {8..15, 24..31}.  Two 16-byte global loads per lane.
__device__ __forceinline__ v16h ld_bfrag(const _Float16* p) {
    v8h lo = *(const v8h*)(p);
    v8h hi = *(const v8h*)(p + 16);
    v16h r;
#pragma unroll
    for (int i = 0; i < 8; ++i) { r[i] = lo[i]; r[i + 8] = hi[i]; }
    return r;
}

__device__ __forceinline__ v8f wmma_f16(v16h a, v16h b, v8f c) {
    return __builtin_amdgcn_wmma_f32_16x16x32_f16(
        /*neg_a=*/false, a, /*neg_b=*/false, b,
        /*c_mod=*/(short)0, c, /*reuse_a=*/false, /*reuse_b=*/false);
}

// ---------------------------------------------------------------------------
// Pre-pass 1: fp32 -> fp16 weight conversion (both weight matrices, 262144 ea)
// ---------------------------------------------------------------------------
__global__ void pe_convert_weights(const float* __restrict__ w1,
                                   const float* __restrict__ w2,
                                   _Float16* __restrict__ w1h,
                                   _Float16* __restrict__ w2h) {
    int id = blockIdx.x * blockDim.x + threadIdx.x;
    if (id < HDIM * CDIM) {
        w1h[id] = (_Float16)w1[id];
        w2h[id] = (_Float16)w2[id];
    }
}

// ---------------------------------------------------------------------------
// Pre-pass 2: mod = silu(c_BD) @ ada_w.T + ada_b   -> (B, 3C) in workspace
// ---------------------------------------------------------------------------
__global__ void pe_mod(const float* __restrict__ c_BD,
                       const float* __restrict__ ada_w,
                       const float* __restrict__ ada_b,
                       float* __restrict__ mod) {
    int id = blockIdx.x * blockDim.x + threadIdx.x;
    if (id >= BATCH * 3 * CDIM) return;
    int b = id / (3 * CDIM);
    int r = id % (3 * CDIM);
    const float* cb = c_BD + b * CDIM;
    const float* wr = ada_w + r * CDIM;
    float acc = ada_b[r];
    for (int k = 0; k < CDIM; ++k) {
        float cv = cb[k];
        float sl = cv / (1.0f + expf(-cv));   // silu
        acc += sl * wr[k];
    }
    mod[id] = acc;
}

// ---------------------------------------------------------------------------
// Main fused kernel: 64 tokens per workgroup, 8 waves (wave32)
// ---------------------------------------------------------------------------
__global__ void __launch_bounds__(256, 2)
pe_main(const float* __restrict__ x,        // (B, L, 3)
        const float* __restrict__ embed_w,  // (NB, C)
        const float* __restrict__ pos_w,    // (128, C)
        const float* __restrict__ bins,     // (NB-1,)
        const float* __restrict__ ln_g,
        const float* __restrict__ ln_b,
        const float* __restrict__ fc1_b,    // (H,)
        const float* __restrict__ fc2_b,    // (C,)
        const float* __restrict__ mod,      // (B, 3C): shift | scale | gate
        const _Float16* __restrict__ w1h,   // (H, C)
        const _Float16* __restrict__ w2h,   // (C, H)
        float* __restrict__ out)            // (B*L*L, C)
{
    extern __shared__ __align__(16) char smem[];
    float*    sRes = (float*)smem;                          // [TM][SR_STRIDE]
    _Float16* sA   = (_Float16*)(smem + SRES_BYTES);        // [TM][SA_STRIDE]
    _Float16* h1c  = (_Float16*)(smem + SRES_BYTES + SA_BYTES); // [TM][H1_STRIDE]

    const int tid  = threadIdx.x;
    const int w    = tid >> 5;        // wave id 0..7
    const int lane = tid & 31;
    const int l16  = lane & 15;
    const int hhi  = lane >> 4;       // 0 or 1

    const int g0   = blockIdx.x * TM;          // first token of tile
    const int bidx = g0 >> 16;                 // token / (L*L); constant per WG
    const float* modb = mod + bidx * 3 * CDIM; // shift at +0, scale +C, gate +2C

    // ---------------- Phase A: token features + adaLN + LayerNorm -----------
    for (int q = 0; q < 8; ++q) {
        const int t   = w + 8 * q;
        const int g   = g0 + t;
        const int rem = g & (SEQL * SEQL - 1);
        const int i   = rem >> 8;
        const int j   = rem & 255;

        const float* xi = x + (bidx * SEQL + i) * 3;
        const float* xj = x + (bidx * SEQL + j) * 3;
        float d = 0.0f;
#pragma unroll
        for (int dd = 0; dd < 3; ++dd) {
            float df = xi[dd] - xj[dd];
            d += df * df;
        }
        int idx = 0;
        for (int kk = 0; kk < NBINS - 1; ++kk) idx += (bins[kk] < d) ? 1 : 0;

        int rel = i - j;
        rel = rel < -64 ? -64 : (rel > 63 ? 63 : rel);
        rel += 64;

        const float* erow = embed_w + idx * CDIM;
        const float* prow = pos_w + rel * CDIM;

        float sv[8];
        float sum = 0.0f, ssq = 0.0f;
#pragma unroll
        for (int u = 0; u < 8; ++u) {
            const int c = lane + 32 * u;
            float v = erow[c] + prow[c];
            float s = v * (1.0f + modb[CDIM + c]) + modb[c];
            sv[u] = s;
            sum += s;
            ssq += s * s;
            sRes[t * SR_STRIDE + c] = s;
        }
#pragma unroll
        for (int off = 16; off >= 1; off >>= 1) {
            sum += __shfl_xor(sum, off, 32);
            ssq += __shfl_xor(ssq, off, 32);
        }
        const float mu  = sum * (1.0f / CDIM);
        const float var = ssq * (1.0f / CDIM) - mu * mu;
        const float rs  = rsqrtf(var + 1e-5f);
#pragma unroll
        for (int u = 0; u < 8; ++u) {
            const int c = lane + 32 * u;
            float h0 = (sv[u] - mu) * rs * ln_g[c] + ln_b[c];
            sA[t * SA_STRIDE + c] = (_Float16)h0;
        }
    }
    __syncthreads();

    // ---------------- Phase B: fused MLP via WMMA ---------------------------
    const v8f vzero = {};
    v8f acc[8];                      // GEMM2 accumulators: nb={2w,2w+1} x mb=0..3
#pragma unroll
    for (int m = 0; m < 8; ++m) acc[m] = vzero;

#pragma unroll 1
    for (int kc = 0; kc < HDIM / KCHUNK; ++kc) {
        const int hk = kc * KCHUNK;

        // ---- GEMM1: h1_chunk = gelu(sA[64x256] @ W1^T[:, hk..hk+128)) ------
        v8f c1[4];
#pragma unroll
        for (int m = 0; m < 4; ++m) c1[m] = vzero;

        const int nrow = hk + w * 16 + l16;        // output channel of H
        const float bias1 = fc1_b[nrow];
        const _Float16* w1row = w1h + nrow * CDIM;
#pragma unroll
        for (int ks = 0; ks < 8; ++ks) {
            v16h bf = ld_bfrag(w1row + ks * 32 + (hhi << 3));
#pragma unroll
            for (int mb = 0; mb < 4; ++mb) {
                const _Float16* ap =
                    sA + (mb * 16 + l16) * SA_STRIDE + ks * 32 + (hhi << 3);
                v16h af = ld_afrag(ap);
                c1[mb] = wmma_f16(af, bf, c1[mb]);
            }
        }

        __syncthreads();   // previous chunk's GEMM2 done reading h1c

        const int ncol = w * 16 + l16;
#pragma unroll
        for (int mb = 0; mb < 4; ++mb) {
#pragma unroll
            for (int r = 0; r < 8; ++r) {
                float xv = c1[mb][r] + bias1;
                float gl = 0.5f * xv * (1.0f + erff(xv * 0.70710678118654752f));
                const int M = mb * 16 + r + (hhi << 3);
                h1c[M * H1_STRIDE + ncol] = (_Float16)gl;
            }
        }
        __syncthreads();

        // ---- GEMM2 partial: acc += h1_chunk[64x128] @ W2^T[hk..hk+128, :] --
        const int n0 = (2 * w) * 16 + l16;
        const _Float16* w2r0 = w2h + n0 * HDIM + hk;
        const _Float16* w2r1 = w2h + (n0 + 16) * HDIM + hk;
#pragma unroll
        for (int ks = 0; ks < 4; ++ks) {
            const int kb = ks * 32 + (hhi << 3);
            v16h b0 = ld_bfrag(w2r0 + kb);
            v16h b1 = ld_bfrag(w2r1 + kb);
#pragma unroll
            for (int mb = 0; mb < 4; ++mb) {
                const _Float16* ap =
                    h1c + (mb * 16 + l16) * H1_STRIDE + ks * 32 + (hhi << 3);
                v16h af = ld_afrag(ap);
                acc[2 * mb + 0] = wmma_f16(af, b0, acc[2 * mb + 0]);
                acc[2 * mb + 1] = wmma_f16(af, b1, acc[2 * mb + 1]);
            }
        }
    }

    // ---------------- Phase C: out = s + gate * (h + fc2_b) -----------------
    const float* gatep = modb + 2 * CDIM;
#pragma unroll
    for (int tc = 0; tc < 2; ++tc) {
        const int N  = (2 * w + tc) * 16 + l16;
        const float gv = gatep[N];
        const float b2 = fc2_b[N];
#pragma unroll
        for (int mb = 0; mb < 4; ++mb) {
#pragma unroll
            for (int r = 0; r < 8; ++r) {
                const int M = mb * 16 + r + (hhi << 3);
                const float res = sRes[M * SR_STRIDE + N];
                out[(size_t)(g0 + M) * CDIM + N] =
                    res + gv * (acc[2 * mb + tc][r] + b2);
            }
        }
    }
}

// ---------------------------------------------------------------------------
// Launcher
// ---------------------------------------------------------------------------
extern "C" void kernel_launch(void* const* d_in, const int* in_sizes, int n_in,
                              void* d_out, int out_size, void* d_ws, size_t ws_size,
                              hipStream_t stream) {
    (void)in_sizes; (void)n_in; (void)out_size; (void)ws_size;

    const float* x       = (const float*)d_in[0];   // (B, L, 3)
    const float* c_BD    = (const float*)d_in[1];   // (B, C)
    const float* embed_w = (const float*)d_in[2];   // (NB, C)
    const float* pos_w   = (const float*)d_in[3];   // (128, C)
    const float* bins    = (const float*)d_in[4];   // (NB-1,)
    const float* ada_w   = (const float*)d_in[5];   // (3C, C)
    const float* ada_b   = (const float*)d_in[6];   // (3C,)
    const float* ln_g    = (const float*)d_in[7];   // (C,)
    const float* ln_b    = (const float*)d_in[8];   // (C,)
    const float* fc1_w   = (const float*)d_in[9];   // (H, C)
    const float* fc1_b   = (const float*)d_in[10];  // (H,)
    const float* fc2_w   = (const float*)d_in[11];  // (C, H)
    const float* fc2_b   = (const float*)d_in[12];  // (C,)
    float* out = (float*)d_out;

    // Workspace layout: mod (B*3C f32) | W1 fp16 | W2 fp16
    float*    mod = (float*)d_ws;
    _Float16* w1h = (_Float16*)((char*)d_ws + 8192);
    _Float16* w2h = w1h + (size_t)HDIM * CDIM;

    // Pre-pass kernels
    pe_convert_weights<<<(HDIM * CDIM + 255) / 256, 256, 0, stream>>>(
        fc1_w, fc2_w, w1h, w2h);
    pe_mod<<<(BATCH * 3 * CDIM + 255) / 256, 256, 0, stream>>>(
        c_BD, ada_w, ada_b, mod);

    // Main fused kernel (needs >64KB dynamic LDS)
    (void)hipFuncSetAttribute((const void*)pe_main,
                              hipFuncAttributeMaxDynamicSharedMemorySize,
                              SMEM_BYTES);
    const int n_tokens = BATCH * SEQL * SEQL;
    pe_main<<<n_tokens / TM, 256, SMEM_BYTES, stream>>>(
        x, embed_w, pos_w, bins, ln_g, ln_b, fc1_b, fc2_b,
        mod, w1h, w2h, out);
}
